// DualPerFreFeatureExtractor_59691455480587
// MI455X (gfx1250) — matrix operand order repaired and checked by reading, not compiled, not run on hardware
//
#include <hip/hip_runtime.h>
#include <hip/hip_bf16.h>
#include <math.h>

// ---------------- problem constants ----------------
#define BB    256
#define PP    256
#define CHN   3
#define NBAND 16
#define DDIM  48
#define SSEQ  256

typedef __bf16 bf16_t;
typedef bf16_t bf16x16 __attribute__((ext_vector_type(16)));
typedef float  f32x8   __attribute__((ext_vector_type(8)));

// gfx1250 async global->LDS (GLOBAL_LOAD_ASYNC_TO_LDS_B32, ASYNCcnt) if the
// toolchain exposes it; falls back to plain LDS stores otherwise.
// Probe result: signature is (AS1 int* gsrc, AS3 int* ldst, imm offset, imm cpol).
#if __has_builtin(__builtin_amdgcn_global_load_async_to_lds_b32) && __has_builtin(__builtin_amdgcn_s_wait_asynccnt)
#define USE_ASYNC_LDS 1
typedef __attribute__((address_space(1))) int* as1_ip;
typedef __attribute__((address_space(3))) int* as3_ip;
#else
#define USE_ASYNC_LDS 0
#endif

__device__ __forceinline__ float gelu_exact(float x){
  return 0.5f * x * (1.0f + erff(x * 0.70710678118654752f));
}
__device__ __forceinline__ float sigmoid_(float x){ return 1.0f / (1.0f + expf(-x)); }

// =====================================================================
// Generic bf16 WMMA GEMM:  C[M,N] = act(A[M,K] @ W[K,N] + bias) + res
//   res indexed as res[(m & resMask)*N + n]  (resMask=P-1 -> pos-enc
//   broadcast, resMask=0x3FFFFFFF -> standard residual). act: 0=none,1=GELU.
// Tile: 128x64 per 256-thread (8-wave) workgroup, BK=32 = one WMMA K-step.
// All launches use M % 128 == 0, so row bounds never apply. Partial K/N
// tiles: zero-fill LDS, barrier, then a compact unconditional load of the
// valid Kr x Nv region (no per-element predicated loads).
// =====================================================================
#define GBM 128
#define GBN 64
#define GBK 32
#define LDSA 36   // element stride (even -> K-pairs stay dword aligned)

__global__ __launch_bounds__(256) void k_gemm(
    const float* __restrict__ A, const float* __restrict__ W,
    const float* __restrict__ bias, const float* __restrict__ res, unsigned resMask,
    float* __restrict__ C, int M, int K, int N, int act)
{
  __shared__ bf16_t sA[GBM * LDSA];   // A tile, row-major [m][k]
  __shared__ bf16_t sW[GBN * LDSA];   // W tile, transposed [n][k]
  const int tid  = threadIdx.x;
  const int lane = tid & 31, wave = tid >> 5;
  const int waveM = wave & 3, waveN = wave >> 2;      // 4x2 wave grid
  const int tileM = blockIdx.y * GBM, tileN = blockIdx.x * GBN;

  f32x8 acc[2][2];
  #pragma unroll
  for (int i = 0; i < 2; i++)
    #pragma unroll
    for (int j = 0; j < 2; j++) acc[i][j] = (f32x8){0,0,0,0,0,0,0,0};

  const int ar = tid >> 1, akh = (tid & 1) * 16;      // A fast-path assignment
  const int agr = tileM + ar;                          // rows always in range
  const int wkw = tid >> 3, wn0 = (tid & 7) * 8;      // W fast-path assignment
  const int Nv = (N - tileN < GBN) ? (N - tileN) : GBN;

  for (int k0 = 0; k0 < K; k0 += GBK) {
    const int Kr = (K - k0 < GBK) ? (K - k0) : GBK;
    const bool partA = (Kr < GBK);
    const bool partW = partA || (Nv < GBN);

    if (partA | partW) {       // zero-fill partial tiles, then barrier
      if (partA) for (int idx = tid; idx < GBM * LDSA; idx += 256) sA[idx] = (bf16_t)0.0f;
      if (partW) for (int idx = tid; idx < GBN * LDSA; idx += 256) sW[idx] = (bf16_t)0.0f;
      __syncthreads();
    }

    // ---- A tile (128 x Kr) ----
    if (!partA) {                               // full: 16 contiguous floats/thread
      bf16_t* dstA = &sA[ar * LDSA + akh];
      const float* Arow = A + (size_t)agr * K + k0 + akh;
      #pragma unroll
      for (int kk = 0; kk < 16; kk++) dstA[kk] = (bf16_t)Arow[kk];
    } else {                                    // compact unconditional loads
      for (int idx = tid; idx < GBM * Kr; idx += 256) {
        int r = idx / Kr, k = idx - r * Kr;
        sA[r * LDSA + k] = (bf16_t)A[(size_t)(tileM + r) * K + k0 + k];
      }
    }
    // ---- W tile (Kr x Nv), stored transposed ----
    if (!partW) {                               // full: 8 contiguous floats/thread
      const float* Wrow = W + (size_t)(k0 + wkw) * N + tileN + wn0;
      #pragma unroll
      for (int j = 0; j < 8; j++) sW[(wn0 + j) * LDSA + wkw] = (bf16_t)Wrow[j];
    } else {                                    // compact unconditional loads
      for (int idx = tid; idx < Kr * Nv; idx += 256) {
        int k = idx / Nv, n = idx - k * Nv;
        sW[n * LDSA + k] = (bf16_t)W[(size_t)(k0 + k) * N + tileN + n];
      }
    }
    __syncthreads();

    // prefetch next A k-slab (gfx1250 global_prefetch_b8)
    if (k0 + GBK < K) __builtin_prefetch(&A[(size_t)agr * K + k0 + GBK], 0, 1);

    // ---- build fragments per ISA 7.12.2 layouts ----
    const int mloc = lane & 15, sel = lane >> 4;
    bf16x16 af[2], bf[2];
    #pragma unroll
    for (int mt = 0; mt < 2; mt++) {
      int row = waveM * 32 + mt * 16 + mloc;
      bf16x16 a;
      #pragma unroll
      for (int j = 0; j < 8; j++) {           // A: K = {0..7|16..23} + sel*8
        int kb = (j < 4 ? 2 * j : 16 + 2 * (j - 4)) + sel * 8;
        a[2 * j]     = sA[row * LDSA + kb];
        a[2 * j + 1] = sA[row * LDSA + kb + 1];
      }
      af[mt] = a;
    }
    #pragma unroll
    for (int nt = 0; nt < 2; nt++) {
      int col = waveN * 32 + nt * 16 + mloc;
      bf16x16 b;
      #pragma unroll
      for (int j = 0; j < 8; j++) {           // B: K = {0..7|16..23} + sel*8
        int kb = (j < 4 ? 2 * j : 16 + 2 * (j - 4)) + sel * 8;
        b[2 * j]     = sW[col * LDSA + kb];
        b[2 * j + 1] = sW[col * LDSA + kb + 1];
      }
      bf[nt] = b;
    }
    #pragma unroll
    for (int mt = 0; mt < 2; mt++)
      #pragma unroll
      for (int nt = 0; nt < 2; nt++)
        acc[mt][nt] = __builtin_amdgcn_wmma_f32_16x16x32_bf16(
            false, af[mt], false, bf[nt], (short)0, acc[mt][nt], false, false);
    __syncthreads();
  }

  // ---- epilogue: bias -> act -> residual -> store ----
  const int nl = lane & 15, mh = (lane >> 4) * 8;
  for (int mt = 0; mt < 2; mt++)
    for (int nt = 0; nt < 2; nt++) {
      int n = tileN + waveN * 32 + nt * 16 + nl;
      if (n >= N) continue;
      #pragma unroll
      for (int j = 0; j < 8; j++) {
        int m = tileM + waveM * 32 + mt * 16 + mh + j;
        if (m >= M) continue;
        float v = acc[mt][nt][j];
        if (bias) v += bias[n];
        if (act == 1) v = gelu_exact(v);
        if (res)  v += res[(size_t)(m & resMask) * N + n];
        C[(size_t)m * N + n] = v;
      }
    }
}

// =====================================================================
// 7-statistic kernel over the band axis: st = [var,mean,max,min,peak,rms,energy]
// =====================================================================
__global__ void k_stats(const float* __restrict__ x, float* __restrict__ st, int R)
{
  int i = blockIdx.x * blockDim.x + threadIdx.x;
  if (i >= R) return;
  const float* p = x + (size_t)i * NBAND;
  float mx = -3.4e38f, mn = 3.4e38f, s = 0.f, s2 = 0.f; int am = 0;
  #pragma unroll
  for (int j = 0; j < NBAND; j++) {
    float a = p[j];
    s += a; s2 += a * a;
    if (a > mx) { mx = a; am = j; }
    if (a < mn) mn = a;
  }
  float mean = s / 16.f;
  float var  = (s2 - 16.f * mean * mean) / 15.f;   // ddof=1
  float* o = st + (size_t)i * 7;
  o[0] = var; o[1] = mean; o[2] = mx; o[3] = mn;
  o[4] = (float)am; o[5] = sqrtf(s2 / 16.f); o[6] = s2;
}

// =====================================================================
// Training-mode BatchNorm over contiguous view (N, C, L), channel axis C.
// =====================================================================
__global__ __launch_bounds__(256) void k_bn_reduce(
    const float* __restrict__ x, float* __restrict__ mv, int N, int C, int L)
{
  int c = blockIdx.x;
  __shared__ float ss[256], ss2[256];
  float s = 0.f, s2 = 0.f;
  for (int n = 0; n < N; n++) {
    const float* base = x + (size_t)n * C * L + (size_t)c * L;
    for (int l = threadIdx.x; l < L; l += 256) {
      float v = base[l];
      s += v; s2 += v * v;
    }
  }
  ss[threadIdx.x] = s; ss2[threadIdx.x] = s2; __syncthreads();
  for (int o = 128; o > 0; o >>= 1) {
    if (threadIdx.x < o) { ss[threadIdx.x] += ss[threadIdx.x + o]; ss2[threadIdx.x] += ss2[threadIdx.x + o]; }
    __syncthreads();
  }
  if (threadIdx.x == 0) {
    float inv = 1.0f / ((float)N * (float)L);
    float mean = ss[0] * inv;
    float var  = ss2[0] * inv - mean * mean;
    mv[2 * c] = mean; mv[2 * c + 1] = rsqrtf(var + 1e-5f);
  }
}

// grid = (cdiv(L,256), C, N); act: 1=gelu 2=relu
__global__ void k_bn_apply(float* __restrict__ x, const float* __restrict__ mv,
                           const float* __restrict__ g, const float* __restrict__ be,
                           int C, int L, int act)
{
  int l = blockIdx.x * blockDim.x + threadIdx.x;
  if (l >= L) return;
  int c = blockIdx.y, n = blockIdx.z;
  size_t i = ((size_t)n * C + c) * L + l;
  float v = (x[i] - mv[2 * c]) * mv[2 * c + 1] * g[c] + be[c];
  if (act == 1) v = gelu_exact(v);
  else if (act == 2) v = fmaxf(v, 0.f);
  x[i] = v;
}

// =====================================================================
// LayerNorm over D=48: one wave32 per row (lane covers col, col+32)
// =====================================================================
__global__ __launch_bounds__(256) void k_ln(
    const float* __restrict__ x, const float* __restrict__ g,
    const float* __restrict__ be, float* __restrict__ o, int M)
{
  int row  = blockIdx.x * 8 + (threadIdx.x >> 5);
  int lane = threadIdx.x & 31;
  if (row >= M) return;
  const float* p = x + (size_t)row * DDIM;
  float a = p[lane];
  float b = (lane < 16) ? p[lane + 32] : 0.f;
  float s = a + b;
  for (int m = 16; m > 0; m >>= 1) s += __shfl_xor(s, m, 32);
  float mean = s / 48.f;
  float da = a - mean, db = (lane < 16) ? (b - mean) : 0.f;
  float s2 = da * da + db * db;
  for (int m = 16; m > 0; m >>= 1) s2 += __shfl_xor(s2, m, 32);
  float r = rsqrtf(s2 / 48.f + 1e-5f);
  float* q = o + (size_t)row * DDIM;
  q[lane] = da * r * g[lane] + be[lane];
  if (lane < 16) q[lane + 32] = db * r * g[lane + 32] + be[lane + 32];
}

// =====================================================================
// Flash attention, dh=8, S=256: one query per thread, K/V staged in LDS,
// single-pass online softmax (exact).
// =====================================================================
__global__ __launch_bounds__(256) void k_attn(
    const float* __restrict__ Q, int ldq, const float* __restrict__ Kp, int ldk,
    const float* __restrict__ V, int ldv, float* __restrict__ O, int ldo, int heads)
{
  int b = blockIdx.x / heads, h = blockIdx.x % heads;
  __shared__ float Ks[SSEQ][8], Vs[SSEQ][8];
  const float* Kb = Kp + (size_t)b * SSEQ * ldk + h * 8;
  const float* Vb = V  + (size_t)b * SSEQ * ldv + h * 8;
  int t = threadIdx.x;
  #pragma unroll
  for (int d = 0; d < 8; d++) {
    Ks[t][d] = Kb[(size_t)t * ldk + d];
    Vs[t][d] = Vb[(size_t)t * ldv + d];
  }
  __syncthreads();
  const float* Qr = Q + ((size_t)b * SSEQ + t) * ldq + h * 8;
  float q[8];
  #pragma unroll
  for (int d = 0; d < 8; d++) q[d] = Qr[d] * 0.35355339059327373f; // 8^-0.5
  float m = -3.4e38f, l = 0.f, o[8] = {0,0,0,0,0,0,0,0};
  for (int k = 0; k < SSEQ; k++) {
    float s = 0.f;
    #pragma unroll
    for (int d = 0; d < 8; d++) s += q[d] * Ks[k][d];
    float mn = fmaxf(m, s);
    float corr = expf(m - mn);
    float pe   = expf(s - mn);
    l = l * corr + pe;
    #pragma unroll
    for (int d = 0; d < 8; d++) o[d] = o[d] * corr + pe * Vs[k][d];
    m = mn;
  }
  float inv = 1.f / l;
  float* Or = O + ((size_t)b * SSEQ + t) * ldo + h * 8;
  #pragma unroll
  for (int d = 0; d < 8; d++) Or[d] = o[d] * inv;
}

// =====================================================================
// Gate apply + positional prep: xt = sigmoid(gate)*x ; m = xt + pos_dec
// =====================================================================
__global__ void k_gate_pos(const float* __restrict__ x, const float* __restrict__ gate,
                           const float* __restrict__ posdec, float* __restrict__ xt,
                           float* __restrict__ mbuf, size_t total)
{
  size_t i = blockIdx.x * (size_t)blockDim.x + threadIdx.x;
  if (i >= total) return;
  size_t bpc = i >> 4;                         // token*3+c index
  float v = sigmoid_(gate[bpc]) * x[i];
  xt[i] = v;
  size_t row = i / DDIM; int col = (int)(i % DDIM);
  mbuf[i] = v + posdec[(row & (PP - 1)) * DDIM + col];
}

// ---- small data-movement / elementwise kernels ----
__global__ void k_dcprep(const float* __restrict__ x, float* __restrict__ o, int total)
{
  int i = blockIdx.x * blockDim.x + threadIdx.x; if (i >= total) return;
  int b = i / (3 * 256); int c = (i / 256) % 3; int p = i % 256;
  o[i] = x[((size_t)b * PP + p) * DDIM + c * NBAND];    // band 0
}
__global__ void k_transpose_y(const float* __restrict__ x, float* __restrict__ y, int total)
{
  int i = blockIdx.x * blockDim.x + threadIdx.x; if (i >= total) return;
  int b = i / (48 * 256); int ch = (i / 256) % 48; int p = i % 256;
  int band = ch / 3, c = ch % 3;
  y[i] = x[((size_t)b * PP + p) * DDIM + c * NBAND + band];
}
__global__ void k_add2(float* __restrict__ o, const float* __restrict__ a,
                       const float* __restrict__ b, size_t total)
{
  size_t i = blockIdx.x * (size_t)blockDim.x + threadIdx.x; if (i >= total) return;
  o[i] = a[i] + b[i];
}

// =====================================================================
// Direct grouped 3x3 conv (SAME), 16x16 spatial, 1 thread/pixel.
// Input channels chunk-staged into 32KB LDS via the gfx1250 async
// global->LDS path (ASYNCcnt) when available.
// =====================================================================
__global__ __launch_bounds__(256) void k_conv3x3(
    const float* __restrict__ in, const float* __restrict__ w,
    const float* __restrict__ bias, float* __restrict__ out,
    int Cin, int Cout, int groups)
{
  __shared__ float ls[32 * 256];
  int b = blockIdx.x, co = blockIdx.y;
  int cpg_in = Cin / groups, cpg_out = Cout / groups;
  int g = co / cpg_out;
  const float* inb = in + ((size_t)b * Cin + g * cpg_in) * 256;
  int px = threadIdx.x, x0 = px & 15, y0 = px >> 4;
  float acc = bias[co];
  const float* wb = w + (size_t)co * cpg_in * 9;
  for (int cb = 0; cb < cpg_in; cb += 32) {
    int cn = cpg_in - cb; if (cn > 32) cn = 32;
    const float* src = inb + (size_t)cb * 256;
#if USE_ASYNC_LDS
    for (int t = threadIdx.x; t < cn * 256; t += 256) {  // cn*256 % 256 == 0 -> EXEC all-1s
      float* gsrc = const_cast<float*>(src + t);
      __builtin_amdgcn_global_load_async_to_lds_b32(
          (as1_ip)gsrc, (as3_ip)(ls + t), 0, 0);
    }
    __builtin_amdgcn_s_wait_asynccnt(0);
#else
    for (int t = threadIdx.x; t < cn * 256; t += 256) ls[t] = src[t];
#endif
    __syncthreads();
    for (int ci = 0; ci < cn; ci++) {
      const float* wc = wb + (size_t)(cb + ci) * 9;
      const float* lc = ls + ci * 256;
      #pragma unroll
      for (int ky = -1; ky <= 1; ky++) {
        int yy = y0 + ky; if (yy < 0 || yy > 15) continue;
        #pragma unroll
        for (int kx = -1; kx <= 1; kx++) {
          int xx = x0 + kx; if (xx < 0 || xx > 15) continue;
          acc += lc[yy * 16 + xx] * wc[(ky + 1) * 3 + (kx + 1)];
        }
      }
    }
    __syncthreads();
  }
  out[((size_t)b * Cout + co) * 256 + px] = acc;
}

// ---- SE gate ----
__global__ void k_hw_mean(const float* __restrict__ yb, float* __restrict__ s)
{
  int b = blockIdx.x; int c = threadIdx.x;
  if (c >= 48) return;
  const float* p = yb + ((size_t)b * 48 + c) * 256;
  float acc = 0.f;
  for (int i = 0; i < 256; i++) acc += p[i];
  s[b * 48 + c] = acc / 256.f;
}
__global__ void k_se_gate(const float* __restrict__ s, const float* __restrict__ w1,
                          const float* __restrict__ w2, float* __restrict__ scl)
{
  int b = blockIdx.x; int t = threadIdx.x;
  __shared__ float sv[48], hv[8];
  if (t < 48) sv[t] = s[b * 48 + t];
  __syncthreads();
  if (t < 8) { float a = 0.f; for (int i = 0; i < 48; i++) a += sv[i] * w1[i * 8 + t]; hv[t] = fmaxf(a, 0.f); }
  __syncthreads();
  if (t < 48) { float a = 0.f; for (int j = 0; j < 8; j++) a += hv[j] * w2[j * 48 + t]; scl[b * 48 + t] = sigmoid_(a); }
}

// ---- final assembly: (B,P,6,16) = concat(patch(+dc@band0), SE-scaled band) ----
__global__ void k_assemble(const float* __restrict__ mbuf, const float* __restrict__ dc,
                           const float* __restrict__ yb, const float* __restrict__ scl,
                           float* __restrict__ out, size_t total)
{
  size_t i = blockIdx.x * (size_t)blockDim.x + threadIdx.x;
  if (i >= total) return;
  int band = (int)(i % 16);
  int c6   = (int)((i / 16) % 6);
  size_t p = (i / 96) % PP;
  size_t b = i / ((size_t)96 * PP);
  float v;
  if (c6 < 3) {
    v = mbuf[(b * PP + p) * DDIM + c6 * NBAND + band];
    if (band == 0) v += dc[(b * 3 + c6) * 256 + p];
  } else {
    int ch = band * 3 + (c6 - 3);
    v = yb[(b * 48 + ch) * 256 + p] * scl[b * 48 + ch];
  }
  out[i] = v;
}

// =====================================================================
// Host orchestration
// =====================================================================
static inline unsigned cdiv(size_t a, size_t b) { return (unsigned)((a + b - 1) / b); }

// param spec indices (d_in index = 2 + spec)
enum {
  S_DC_W = 0, S_DC_B = 1, S_B0 = 2,           // b{i} base = 2 + 8*i: w1,b1,g1,be1,w2,b2,g2,be2
  S_SE_W1 = 26, S_SE_W2 = 27, S_EMB_W = 28, S_EMB_B = 29,
  S_POS_ENC = 30, S_POS_DEC = 31,
  S_E_LN1G = 32, S_E_LN1B = 33, S_E_QKV = 34, S_E_OW = 35, S_E_OB = 36,
  S_E_LN2G = 37, S_E_LN2B = 38, S_E_FF1W = 39, S_E_FF1B = 40, S_E_FF2W = 41, S_E_FF2B = 42,
  S_D_LN1G = 43, S_D_LN1B = 44, S_D_SAQKV = 45, S_D_SAOW = 46, S_D_SAOB = 47,
  S_D_LN2G = 48, S_D_LN2B = 49, S_D_CAQ = 50, S_D_CAKV = 51, S_D_CAOW = 52, S_D_CAOB = 53,
  S_D_LN3G = 54, S_D_LN3B = 55, S_D_FF1W = 56, S_D_FF1B = 57, S_D_FF2W = 58, S_D_FF2B = 59,
  S_ST = 60, S_TP = 70, S_PA = 80            // mlp3: (w1,b1,bn1g,bn1b,w2,b2,bn2g,bn2b,w3,b3)
};

#define RES_FULL 0x3FFFFFFFu   // identity mask (m < 2^21 for all launches)

static void run_one(const float* x, float* out, float* ws, void* const* d_in, hipStream_t st_)
{
  const int R  = BB * PP * CHN;   // 196608 mlp3 rows (mult of 128)
  const int M2 = BB * PP;         // 65536 token rows (mult of 128)
  auto prm = [&](int i) { return (const float*)d_in[2 + i]; };

  // ---- workspace layout (floats); conv buffers alias dead MLP3 buffers ----
  size_t off = 0;
  auto alloc = [&](size_t n) { float* r = ws + off; off += n; return r; };
  float* ST_  = alloc((size_t)R * 7);
  float* T1   = alloc((size_t)R * 128);
  float* T2   = alloc((size_t)R * 128);
  float* T3   = alloc((size_t)R * 128);
  float* GATE = alloc((size_t)R);
  float* MV   = alloc(1024);
  float* XT   = alloc((size_t)M2 * DDIM);
  float* MBUF = alloc((size_t)M2 * DDIM);
  float* XE   = alloc((size_t)M2 * DDIM);
  float* XN   = alloc((size_t)M2 * DDIM);
  float* QKV  = alloc((size_t)M2 * 192);
  float* ATT  = alloc((size_t)M2 * 64);
  float* DCIN = alloc((size_t)BB * 3 * 256);
  float* DCO  = alloc((size_t)BB * 3 * 256);
  float* SV   = alloc((size_t)BB * 48);
  float* SCL  = alloc((size_t)BB * 48);
  float* Y  = T1;                                   // aliases (mlp3 chain is done)
  float* YB = T1 + (size_t)BB * 48 * 256;
  float* CM = T2;
  float* CB = T3;

  auto gemm = [&](const float* Ap, int M, int K, const float* Wp, int N,
                  const float* bias, const float* res, unsigned resMask, float* Cp, int act) {
    dim3 g(cdiv(N, GBN), cdiv(M, GBM));
    k_gemm<<<g, 256, 0, st_>>>(Ap, Wp, bias, res, resMask, Cp, M, K, N, act);
  };
  auto bn = [&](float* buf, int C, int L, const float* g_, const float* be_, int act) {
    k_bn_reduce<<<C, 256, 0, st_>>>(buf, MV, BB, C, L);
    dim3 g(cdiv(L, 256), C, BB);
    k_bn_apply<<<g, 256, 0, st_>>>(buf, MV, g_, be_, C, L, act);
  };
  auto ln = [&](const float* in, int gi, int bi, float* o) {
    k_ln<<<M2 / 8, 256, 0, st_>>>(in, prm(gi), prm(bi), o, M2);
  };

  // ---- stats + gate MLPs ----
  k_stats<<<cdiv(R, 256), 256, 0, st_>>>(x, ST_, R);
  // st chain: 7 -> 32 -> 64 -> 128
  gemm(ST_, R, 7,  prm(S_ST + 0), 32, prm(S_ST + 1), nullptr, 0, T1, 0);
  bn(T1, PP, 3 * 32, prm(S_ST + 2), prm(S_ST + 3), 1);
  gemm(T1, R, 32, prm(S_ST + 4), 64, prm(S_ST + 5), nullptr, 0, T2, 0);
  bn(T2, PP, 3 * 64, prm(S_ST + 6), prm(S_ST + 7), 1);
  gemm(T2, R, 64, prm(S_ST + 8), 128, prm(S_ST + 9), nullptr, 0, T3, 0);
  // tp chain: 16 -> 64 -> 128 -> 128, fused add into T3 (feat = st + tp)
  gemm(x, R, 16, prm(S_TP + 0), 64, prm(S_TP + 1), nullptr, 0, T1, 0);
  bn(T1, PP, 3 * 64, prm(S_TP + 2), prm(S_TP + 3), 1);
  gemm(T1, R, 64, prm(S_TP + 4), 128, prm(S_TP + 5), nullptr, 0, T2, 0);
  bn(T2, PP, 3 * 128, prm(S_TP + 6), prm(S_TP + 7), 1);
  gemm(T2, R, 128, prm(S_TP + 8), 128, prm(S_TP + 9), T3, RES_FULL, T3, 0);
  // pa chain: 128 -> 64 -> 32 -> 1
  gemm(T3, R, 128, prm(S_PA + 0), 64, prm(S_PA + 1), nullptr, 0, T1, 0);
  bn(T1, PP, 3 * 64, prm(S_PA + 2), prm(S_PA + 3), 1);
  gemm(T1, R, 64, prm(S_PA + 4), 32, prm(S_PA + 5), nullptr, 0, T2, 0);
  bn(T2, PP, 3 * 32, prm(S_PA + 6), prm(S_PA + 7), 1);
  gemm(T2, R, 32, prm(S_PA + 8), 1, prm(S_PA + 9), nullptr, 0, GATE, 0);

  // ---- transformer encoder ----
  size_t tok = (size_t)M2 * DDIM;
  k_gate_pos<<<cdiv(tok, 256), 256, 0, st_>>>(x, GATE, prm(S_POS_DEC), XT, MBUF, tok);
  gemm(XT, M2, 48, prm(S_EMB_W), 48, prm(S_EMB_B), prm(S_POS_ENC), PP - 1, XE, 0);
  ln(XE, S_E_LN1G, S_E_LN1B, XN);
  gemm(XN, M2, 48, prm(S_E_QKV), 192, nullptr, nullptr, 0, QKV, 0);
  k_attn<<<BB * 8, 256, 0, st_>>>(QKV, 192, QKV + 64, 192, QKV + 128, 192, ATT, 64, 8);
  gemm(ATT, M2, 64, prm(S_E_OW), 48, prm(S_E_OB), XE, RES_FULL, XE, 0);
  ln(XE, S_E_LN2G, S_E_LN2B, XN);
  gemm(XN, M2, 48, prm(S_E_FF1W), 48, prm(S_E_FF1B), nullptr, 0, ATT, 1);   // GELU, scratch
  gemm(ATT, M2, 48, prm(S_E_FF2W), 48, prm(S_E_FF2B), XE, RES_FULL, XE, 0);

  // ---- transformer decoder ----
  ln(MBUF, S_D_LN1G, S_D_LN1B, XN);
  gemm(XN, M2, 48, prm(S_D_SAQKV), 96, nullptr, nullptr, 0, QKV, 0);
  k_attn<<<BB * 4, 256, 0, st_>>>(QKV, 96, QKV + 32, 96, QKV + 64, 96, ATT, 32, 4);
  gemm(ATT, M2, 32, prm(S_D_SAOW), 48, prm(S_D_SAOB), MBUF, RES_FULL, MBUF, 0);
  ln(MBUF, S_D_LN2G, S_D_LN2B, XN);
  gemm(XN, M2, 48, prm(S_D_CAQ), 32, nullptr, nullptr, 0, QKV, 0);          // Q (ld 32)
  float* KVb = QKV + (size_t)M2 * 32;
  gemm(XE, M2, 48, prm(S_D_CAKV), 64, nullptr, nullptr, 0, KVb, 0);         // K|V (ld 64)
  k_attn<<<BB * 4, 256, 0, st_>>>(QKV, 32, KVb, 64, KVb + 32, 64, ATT, 32, 4);
  gemm(ATT, M2, 32, prm(S_D_CAOW), 48, prm(S_D_CAOB), MBUF, RES_FULL, MBUF, 0);
  ln(MBUF, S_D_LN3G, S_D_LN3B, XN);
  gemm(XN, M2, 48, prm(S_D_FF1W), 8, prm(S_D_FF1B), nullptr, 0, ATT, 1);    // GELU, ld 8
  gemm(ATT, M2, 8, prm(S_D_FF2W), 48, prm(S_D_FF2B), MBUF, RES_FULL, MBUF, 0);

  // ---- DC branch ----
  k_dcprep<<<cdiv((size_t)BB * 3 * 256, 256), 256, 0, st_>>>(x, DCIN, BB * 3 * 256);
  k_conv3x3<<<dim3(BB, 3), 256, 0, st_>>>(DCIN, prm(S_DC_W), prm(S_DC_B), DCO, 3, 3, 1);

  // ---- band conv branches + identity ----
  k_transpose_y<<<cdiv((size_t)BB * 48 * 256, 256), 256, 0, st_>>>(x, Y, BB * 48 * 256);
  const int GRP[3] = {16, 4, 1};
  for (int i = 0; i < 3; i++) {
    int base = S_B0 + 8 * i;
    k_conv3x3<<<dim3(BB, 128), 256, 0, st_>>>(Y, prm(base + 0), prm(base + 1), CM, 48, 128, GRP[i]);
    k_bn_reduce<<<128, 256, 0, st_>>>(CM, MV, BB, 128, 256);
    { dim3 g(1, 128, BB); k_bn_apply<<<g, 256, 0, st_>>>(CM, MV, prm(base + 2), prm(base + 3), 128, 256, 2); }
    k_conv3x3<<<dim3(BB, 48), 256, 0, st_>>>(CM, prm(base + 4), prm(base + 5), CB, 128, 48, GRP[i]);
    k_bn_reduce<<<48, 256, 0, st_>>>(CB, MV, BB, 48, 256);
    { dim3 g(1, 48, BB); k_bn_apply<<<g, 256, 0, st_>>>(CB, MV, prm(base + 6), prm(base + 7), 48, 256, 2); }
    k_add2<<<cdiv((size_t)BB * 48 * 256, 256), 256, 0, st_>>>(
        YB, (i == 0) ? Y : YB, CB, (size_t)BB * 48 * 256);
  }
  // ---- SE gate + assembly ----
  k_hw_mean<<<BB, 64, 0, st_>>>(YB, SV);
  k_se_gate<<<BB, 64, 0, st_>>>(SV, prm(S_SE_W1), prm(S_SE_W2), SCL);
  size_t total = (size_t)BB * PP * 96;
  k_assemble<<<cdiv(total, 256), 256, 0, st_>>>(MBUF, DCO, YB, SCL, out, total);
}

extern "C" void kernel_launch(void* const* d_in, const int* in_sizes, int n_in,
                              void* d_out, int out_size, void* d_ws, size_t ws_size,
                              hipStream_t stream)
{
  (void)in_sizes; (void)n_in; (void)out_size; (void)ws_size;
  const float* t0 = (const float*)d_in[0];
  const float* t1 = (const float*)d_in[1];
  float* out = (float*)d_out;
  float* ws  = (float*)d_ws;
  run_one(t0, out,                         ws, d_in, stream);
  run_one(t1, out + (size_t)BB * PP * 96,  ws, d_in, stream);
}
